// RefineModel_30897994728082
// MI455X (gfx1250) — matrix operand order repaired and checked by reading, not compiled, not run on hardware
//
#include <hip/hip_runtime.h>
#include <hip/hip_bf16.h>

// ---------------------------------------------------------------------------
// RefineModel on gfx1250: conv1 (VALU) -> conv2/conv3 (WMMA f16 implicit GEMM,
// fp32 accumulate, fused BN+ReLU, 1x4 register-blocked tiles) -> conv4 (VALU).
// Intermediates in f16 NHWC.
// ---------------------------------------------------------------------------

typedef __attribute__((ext_vector_type(16))) _Float16 v16h;
typedef __attribute__((ext_vector_type(8)))  _Float16 v8h;
typedef __attribute__((ext_vector_type(8)))  float    v8f;

#define BSZ   2048
#define WDIM  14          // keypoints (spatial W)
#define EPSV  1e-5f

// ---------------------------------------------------------------------------
// BN fold: scale = g*rsqrt(v+eps), bias = b - m*scale
// ---------------------------------------------------------------------------
__global__ void bnprep_kernel(const float* __restrict__ g, const float* __restrict__ b,
                              const float* __restrict__ m, const float* __restrict__ v,
                              float* __restrict__ sc, float* __restrict__ bi, int n) {
    int i = blockIdx.x * blockDim.x + threadIdx.x;
    if (i < n) {
        float s = g[i] * rsqrtf(v[i] + EPSV);
        sc[i] = s;
        bi[i] = b[i] - m[i] * s;
    }
}

// ---------------------------------------------------------------------------
// Weight transpose OIHW fp32 -> [N][(kh*3+kw)*CIN + c] f16
// ---------------------------------------------------------------------------
template<int CIN, int COUT>
__global__ void wtrans_kernel(const float* __restrict__ w, _Float16* __restrict__ wt) {
    int i = blockIdx.x * blockDim.x + threadIdx.x;
    constexpr int TOTAL = COUT * CIN * 9;
    if (i >= TOTAL) return;
    int kw = i % 3;
    int kh = (i / 3) % 3;
    int c  = (i / 9) % CIN;
    int n  = i / (9 * CIN);
    wt[(size_t)n * (9 * CIN) + (kh * 3 + kw) * CIN + c] = (_Float16)w[i];
}

// ---------------------------------------------------------------------------
// conv1: [B,2,64,14] (ch0 = x, ch1 = 10*sigma), 3x3 stride(2,1) pad(1,1),
// BN1+ReLU, write f16 NHWC h1 = [B,32,14,64]
// ---------------------------------------------------------------------------
__global__ __launch_bounds__(256)
void conv1_kernel(const float* __restrict__ x, const float* __restrict__ sig,
                  const float* __restrict__ w1, const float* __restrict__ sc,
                  const float* __restrict__ bi, _Float16* __restrict__ h1) {
    int i  = blockIdx.x * blockDim.x + threadIdx.x;   // exact grid
    int co = i & 63;
    int ow = (i >> 6) % WDIM;
    int oh = (i / (64 * WDIM)) % 32;
    int b  = i / (64 * WDIM * 32);

    const float* xb = x   + (size_t)b * 64 * WDIM;
    const float* sb = sig + (size_t)b * 64 * WDIM;

    float s = 0.f;
    #pragma unroll
    for (int kh = 0; kh < 3; ++kh) {
        int ih = oh * 2 - 1 + kh;
        if ((unsigned)ih >= 64u) continue;
        #pragma unroll
        for (int kw = 0; kw < 3; ++kw) {
            int iw = ow - 1 + kw;
            if ((unsigned)iw >= (unsigned)WDIM) continue;
            float i0 = xb[ih * WDIM + iw];
            float i1 = sb[ih * WDIM + iw] * 10.f;
            s += i0 * w1[((co * 2 + 0) * 3 + kh) * 3 + kw];
            s += i1 * w1[((co * 2 + 1) * 3 + kh) * 3 + kw];
        }
    }
    float r = s * sc[co] + bi[co];
    r = r > 0.f ? r : 0.f;
    h1[(((size_t)b * 32 + oh) * WDIM + ow) * 64 + co] = (_Float16)r;
}

// ---------------------------------------------------------------------------
// Implicit-GEMM conv (3x3, stride(2,1), pad(1,1)) with WMMA f32_16x16x32_f16.
// One wave owns one 16(M) x NB*16(N) block: the A fragment is loaded once per
// 32-wide K step and reused across NB WMMAs (NB independent accumulator
// chains keep the XDL pipe busy).  K-loop is blk(9 taps) x cb(CIN/32) so
// padding checks and tap geometry are computed once per tap.  Fused BN+ReLU
// epilogue, f16 NHWC output.
// ---------------------------------------------------------------------------
template<int CIN, int HIN, int HOUT, int COUT, int NB>
__global__ __launch_bounds__(256)
void conv_wmma_kernel(const _Float16* __restrict__ in,   // [B, HIN, 14, CIN]
                      const _Float16* __restrict__ wt,   // [COUT, 9*CIN]
                      const float* __restrict__ bnscale, // [COUT]
                      const float* __restrict__ bnbias,  // [COUT]
                      _Float16* __restrict__ out)        // [B, HOUT, 14, COUT]
{
    constexpr int KDIM = 9 * CIN;
    constexpr int MT   = (HOUT * WDIM) / 16;   // M tiles per batch element
    constexpr int NTG  = COUT / (16 * NB);     // N tile-groups
    constexpr int CB   = CIN / 32;             // 32-wide K steps per tap

    const int wave = blockIdx.x * (blockDim.x >> 5) + (threadIdx.x >> 5);
    const int lane = threadIdx.x & 31;
    const int lr   = lane & 15;                // A row / B column within tile
    const int half = lane >> 4;                // K half selector (ISA layout)

    const int ntg = wave % NTG;
    const int mt  = (wave / NTG) % MT;
    const int b   = wave / (NTG * MT);

    // A-side geometry: output position for this lane's matrix row
    const int p  = mt * 16 + lr;
    const int oh = p / WDIM;
    const int ow = p % WDIM;

    const _Float16* __restrict__ inb = in + (size_t)b * HIN * WDIM * CIN;

    // B row pointers, one per N sub-tile (lane's weight row, K contiguous)
    const _Float16* wrow[NB];
    #pragma unroll
    for (int nb = 0; nb < NB; ++nb)
        wrow[nb] = wt + (size_t)((ntg * NB + nb) * 16 + lr) * KDIM;

    v8f acc[NB] = {};

    for (int blk = 0; blk < 9; ++blk) {        // 3x3 taps
        const int kh = blk / 3, kw = blk % 3;
        const int ih = oh * 2 - 1 + kh;
        const int iw = ow - 1 + kw;
        const bool valid = ((unsigned)ih < (unsigned)HIN) &
                           ((unsigned)iw < (unsigned)WDIM);
        const _Float16* ap =
            inb + ((size_t)(valid ? ih : 0) * WDIM + (valid ? iw : 0)) * CIN + half * 8;

        #pragma unroll
        for (int cb = 0; cb < CB; ++cb) {
            const int k0 = blk * CIN + cb * 32;

            // ---- A fragment: lane holds K = k0+half*8+[0..7] and +16 ----
            v8h a_lo = {}, a_hi = {};
            if (valid) {
                a_lo = *(const v8h*)(ap + cb * 32);
                a_hi = *(const v8h*)(ap + cb * 32 + 16);
            }
            union { v16h v; v8h h[2]; } ua;
            ua.h[0] = a_lo; ua.h[1] = a_hi;

            // ---- NB WMMAs sharing this A fragment ----
            #pragma unroll
            for (int nb = 0; nb < NB; ++nb) {
                const _Float16* bp = wrow[nb] + k0 + half * 8;
                union { v16h v; v8h h[2]; } ub;
                ub.h[0] = *(const v8h*)(bp);
                ub.h[1] = *(const v8h*)(bp + 16);
                acc[nb] = __builtin_amdgcn_wmma_f32_16x16x32_f16(
                              /*neg_a=*/false, ua.v, /*neg_b=*/false, ub.v,
                              /*c_mod=*/(short)0, acc[nb],
                              /*reuse_a=*/false, /*reuse_b=*/false);
            }
        }
    }

    // Epilogue: D layout -> VGPR r, lanes[0:16) = M=r, lanes[16:32) = M=r+8
    #pragma unroll
    for (int nb = 0; nb < NB; ++nb) {
        const int   n  = (ntg * NB + nb) * 16 + lr;
        const float sc = bnscale[n];
        const float bi = bnbias[n];
        _Float16* __restrict__ ob = out + (size_t)b * HOUT * WDIM * COUT + n;
        #pragma unroll
        for (int r = 0; r < 8; ++r) {
            int pp  = mt * 16 + r + half * 8;
            int o_h = pp / WDIM, o_w = pp % WDIM;
            float v = acc[nb][r] * sc + bi;
            v = v > 0.f ? v : 0.f;
            ob[((size_t)o_h * WDIM + o_w) * COUT] = (_Float16)v;
        }
    }
}

// ---------------------------------------------------------------------------
// conv4: temporal merge.  h3 [B,8,14,256] f16  x  w4 [2,256,8,1] fp32
//   -> out [B,14,2] fp32
// ---------------------------------------------------------------------------
__global__ __launch_bounds__(256)
void conv4_kernel(const _Float16* __restrict__ h3, const float* __restrict__ w4,
                  float* __restrict__ out) {
    int i = blockIdx.x * blockDim.x + threadIdx.x;
    if (i >= BSZ * WDIM * 2) return;
    int j  = i & 1;
    int ow = (i >> 1) % WDIM;
    int b  = i / (2 * WDIM);

    const float* wj = w4 + (size_t)j * 256 * 8;   // [c][fh]
    float s = 0.f;
    for (int fh = 0; fh < 8; ++fh) {
        const _Float16* hp = h3 + (((size_t)b * 8 + fh) * WDIM + ow) * 256;
        #pragma unroll 8
        for (int c = 0; c < 256; ++c)
            s += (float)hp[c] * wj[c * 8 + fh];
    }
    out[i] = s;   // out[b][ow][j]
}

// ---------------------------------------------------------------------------
extern "C" void kernel_launch(void* const* d_in, const int* in_sizes, int n_in,
                              void* d_out, int out_size, void* d_ws, size_t ws_size,
                              hipStream_t stream) {
    // setup_inputs() dict order
    const float* x   = (const float*)d_in[0];
    const float* sig = (const float*)d_in[1];
    const float* w1  = (const float*)d_in[2];
    const float* w2  = (const float*)d_in[3];
    const float* w3  = (const float*)d_in[4];
    const float* w4  = (const float*)d_in[5];
    const float* g1  = (const float*)d_in[6];
    const float* b1  = (const float*)d_in[7];
    const float* m1  = (const float*)d_in[8];
    const float* v1  = (const float*)d_in[9];
    const float* g2  = (const float*)d_in[10];
    const float* b2  = (const float*)d_in[11];
    const float* m2  = (const float*)d_in[12];
    const float* v2  = (const float*)d_in[13];
    const float* g3  = (const float*)d_in[14];
    const float* b3  = (const float*)d_in[15];
    const float* m3  = (const float*)d_in[16];
    const float* v3  = (const float*)d_in[17];
    float* out = (float*)d_out;

    // ---- workspace carve (256B aligned) ----
    const size_t ACT_BYTES = (size_t)BSZ * 28672 * sizeof(_Float16);  // 117,440,512
    char* ws = (char*)d_ws;
    size_t off = 0;
    auto carve = [&](size_t bytes) {
        char* p = ws + off;
        off += (bytes + 255) & ~(size_t)255;
        return p;
    };
    _Float16* bufA = (_Float16*)carve(ACT_BYTES);   // h1, later reused as h3
    _Float16* bufB = (_Float16*)carve(ACT_BYTES);   // h2
    _Float16* w2t  = (_Float16*)carve((size_t)128 * 576  * sizeof(_Float16));
    _Float16* w3t  = (_Float16*)carve((size_t)256 * 1152 * sizeof(_Float16));
    float* sc1 = (float*)carve(64  * sizeof(float));
    float* bi1 = (float*)carve(64  * sizeof(float));
    float* sc2 = (float*)carve(128 * sizeof(float));
    float* bi2 = (float*)carve(128 * sizeof(float));
    float* sc3 = (float*)carve(256 * sizeof(float));
    float* bi3 = (float*)carve(256 * sizeof(float));

    // ---- BN folds ----
    bnprep_kernel<<<1, 64,  0, stream>>>(g1, b1, m1, v1, sc1, bi1, 64);
    bnprep_kernel<<<1, 128, 0, stream>>>(g2, b2, m2, v2, sc2, bi2, 128);
    bnprep_kernel<<<1, 256, 0, stream>>>(g3, b3, m3, v3, sc3, bi3, 256);

    // ---- weight transposes (OIHW fp32 -> [N][K] f16, K = (kh,kw,c)) ----
    wtrans_kernel<64, 128><<<(128 * 64 * 9 + 255) / 256, 256, 0, stream>>>(w2, w2t);
    wtrans_kernel<128, 256><<<(256 * 128 * 9 + 255) / 256, 256, 0, stream>>>(w3, w3t);

    // ---- conv1: 58,720,256 threads (exact grid) ----
    conv1_kernel<<<(BSZ * 32 * WDIM * 64) / 256, 256, 0, stream>>>(x, sig, w1, sc1, bi1, bufA);

    // ---- conv2 WMMA: B * MT(14) * NTG(2) = 57,344 waves, 8 waves/block ----
    conv_wmma_kernel<64, 32, 16, 128, 4><<<(BSZ * 14 * 2) / 8, 256, 0, stream>>>(
        bufA, w2t, sc2, bi2, bufB);

    // ---- conv3 WMMA: B * MT(7) * NTG(4) = 57,344 waves; h3 reuses bufA ----
    conv_wmma_kernel<128, 16, 8, 256, 4><<<(BSZ * 7 * 4) / 8, 256, 0, stream>>>(
        bufB, w3t, sc3, bi3, bufA);

    // ---- conv4 ----
    conv4_kernel<<<(BSZ * WDIM * 2 + 255) / 256, 256, 0, stream>>>(bufA, w4, out);

    (void)in_sizes; (void)n_in; (void)out_size; (void)ws_size;
}
